// Block_88021059764449
// MI455X (gfx1250) — compile-verified, matrix operand order checked by reading, not run on hardware
//
#include <hip/hip_runtime.h>
#include <math.h>

typedef __bf16 bf16_t;
typedef __attribute__((ext_vector_type(16))) __bf16 bf16x16;
typedef __attribute__((ext_vector_type(8)))  __bf16 bf16x8;
typedef __attribute__((ext_vector_type(8)))  float  v8f;

constexpr int kE = 1024;
constexpr int kT = 2048;
constexpr int kB = 2;
constexpr int kH = 16;
constexpr int kD = 64;
constexpr int kRows = kB * kT;   // 4096

// ---------- CDNA5 async global->LDS path (guarded; fallback compiles) -------

#if __has_builtin(__builtin_amdgcn_global_load_async_to_lds_b128)
#define HAVE_ASYNC_LDS 1
#else
#define HAVE_ASYNC_LDS 0
#endif

#if HAVE_ASYNC_LDS
// Param types per hipcc diagnostic: (int4 AS1*, int4 AS3*, Ii offset, Ii cpol)
typedef int v4i_raw __attribute__((__vector_size__(16)));
typedef __attribute__((address_space(1))) v4i_raw* gptr_b128;
typedef __attribute__((address_space(3))) v4i_raw* lptr_b128;

template <int OFF>
__device__ __forceinline__ void async_b128(const bf16_t* g, bf16_t* l) {
  // IOFFSET applies to both global and LDS addresses (ISA §15.18.3):
  // LDS[lds_addr + OFF] = MEM[gaddr + OFF]
  __builtin_amdgcn_global_load_async_to_lds_b128(
      (gptr_b128)(void*)(const_cast<bf16_t*>(g)), (lptr_b128)(void*)l, OFF, 0);
}
#if __has_builtin(__builtin_amdgcn_s_wait_asynccnt)
#define WAIT_ASYNC(n) __builtin_amdgcn_s_wait_asynccnt(n)
#else
#define WAIT_ASYNC(n) asm volatile("s_wait_asynccnt %0" ::"n"(n) : "memory")
#endif
#endif

// ---------- helpers ----------

__device__ __forceinline__ bf16_t f2bf(float f) {
  unsigned u = __builtin_bit_cast(unsigned, f);
  unsigned r = u + 0x7FFFu + ((u >> 16) & 1u);      // round to nearest even
  unsigned short h = (unsigned short)(r >> 16);
  return __builtin_bit_cast(bf16_t, h);
}

__device__ __forceinline__ v8f wmma_bf16(bf16x16 a, bf16x16 b, v8f c) {
  return __builtin_amdgcn_wmma_f32_16x16x32_bf16(false, a, false, b, (short)0, c,
                                                 false, false);
}

// Load a 16x32 bf16 operand fragment (A-layout; B operand is loaded from the
// transposed matrix so the layout is identical).
// Lane l: row m = l&15, khalf = l>>4. Elements 0..7 -> K = kh*8+0..7,
// elements 8..15 -> K = kh*8+16..23  (per CDNA5 ISA 16-bit A-matrix layout).
__device__ __forceinline__ bf16x16 frag_ld(const bf16_t* base, int stride) {
  int lane = threadIdx.x & 31;
  int m = lane & 15, kh = lane >> 4;
  const bf16_t* p = base + m * stride + kh * 8;
  bf16x8 lo = *(const bf16x8*)(p);
  bf16x8 hi = *(const bf16x8*)(p + 16);
  bf16x16 r;
#pragma unroll
  for (int i = 0; i < 8; ++i) { r[i] = lo[i]; r[i + 8] = hi[i]; }
  return r;
}

// ---------- weight convert + transpose: w[K,N] f32 -> wt[N,K] bf16 ----------

__global__ __launch_bounds__(256) void convt_kernel(const float* __restrict__ w,
                                                    bf16_t* __restrict__ wt,
                                                    int K, int N) {
  __shared__ float tile[32][33];
  int kb = blockIdx.x * 32, nb = blockIdx.y * 32;
  int tx = threadIdx.x & 31, ty = threadIdx.x >> 5;   // ty 0..7
#pragma unroll
  for (int i = ty; i < 32; i += 8)
    tile[i][tx] = w[(size_t)(kb + i) * N + nb + tx];
  __syncthreads();
#pragma unroll
  for (int i = ty; i < 32; i += 8)
    wt[(size_t)(nb + i) * K + kb + tx] = f2bf(tile[tx][i]);
}

// ---------- layernorm: x[R,1024] f32 -> out[R,1024] bf16 ----------

__global__ __launch_bounds__(256) void ln_bf16_kernel(const float* __restrict__ x,
                                                      const float* __restrict__ w,
                                                      const float* __restrict__ b,
                                                      bf16_t* __restrict__ out) {
  int row = blockIdx.x;
  int tid = threadIdx.x;
  __shared__ float rsum[8], rsum2[8], stats[2];
  float4 xv = ((const float4*)(x + (size_t)row * kE))[tid];
  float s  = xv.x + xv.y + xv.z + xv.w;
  float s2 = xv.x * xv.x + xv.y * xv.y + xv.z * xv.z + xv.w * xv.w;
#pragma unroll
  for (int m = 16; m; m >>= 1) {
    s  += __shfl_xor(s,  m, 32);
    s2 += __shfl_xor(s2, m, 32);
  }
  if ((tid & 31) == 0) { rsum[tid >> 5] = s; rsum2[tid >> 5] = s2; }
  __syncthreads();
  if (tid == 0) {
    float a = 0.f, c = 0.f;
#pragma unroll
    for (int i = 0; i < 8; ++i) { a += rsum[i]; c += rsum2[i]; }
    float mu = a * (1.0f / kE);
    stats[0] = mu;
    stats[1] = rsqrtf(c * (1.0f / kE) - mu * mu + 1e-5f);
  }
  __syncthreads();
  float mu = stats[0], rs = stats[1];
  float4 wv = ((const float4*)w)[tid];
  float4 bv = ((const float4*)b)[tid];
  size_t o = (size_t)row * kE + tid * 4;
  out[o + 0] = f2bf((xv.x - mu) * rs * wv.x + bv.x);
  out[o + 1] = f2bf((xv.y - mu) * rs * wv.y + bv.y);
  out[o + 2] = f2bf((xv.z - mu) * rs * wv.z + bv.z);
  out[o + 3] = f2bf((xv.w - mu) * rs * wv.w + bv.w);
}

// ---------- GEMM: C[M,N] = A[M,K](bf16) * Bt[N,K]^T(bf16) + bias, epilogues --

// modes
#define GM_F32   0   // outf = acc + bias
#define GM_RESID 1   // outf = acc + bias + resid
#define GM_BF16  2   // outb = acc + bias
#define GM_GELU  3   // outb = gelu(acc + bias)
#define GM_QKV   4   // scatter q (scaled), k, vT as bf16

__global__ __launch_bounds__(256) void gemm_bf16_kernel(
    const bf16_t* __restrict__ A, const bf16_t* __restrict__ Bt,
    const float* __restrict__ bias, const float* __restrict__ resid,
    float* __restrict__ outf, bf16_t* __restrict__ outb,
    bf16_t* __restrict__ qo, bf16_t* __restrict__ ko, bf16_t* __restrict__ vto,
    int M, int N, int K, int mode) {
  constexpr int BM = 128, BN = 128, BK = 32;
#if HAVE_ASYNC_LDS
  __shared__ __align__(16) bf16_t As[2][BM * BK];
  __shared__ __align__(16) bf16_t Bs[2][BN * BK];
#else
  __shared__ __align__(16) bf16_t As[1][BM * BK];
  __shared__ __align__(16) bf16_t Bs[1][BN * BK];
#endif

  int tid = threadIdx.x;
  int wave = tid >> 5;
  int lane = tid & 31;
  int wm = wave & 1;        // 2 waves along M (64 rows each)
  int wn = wave >> 1;       // 4 waves along N (32 cols each)
  int blockN0 = blockIdx.x * BN;
  int blockM0 = blockIdx.y * BM;

  v8f acc[4][2];
#pragma unroll
  for (int i = 0; i < 4; ++i)
#pragma unroll
    for (int j = 0; j < 2; ++j) acc[i][j] = {};

  int la_row = tid >> 1;            // 0..127
  int la_col = (tid & 1) * 16;      // 0 or 16

#if HAVE_ASYNC_LDS
  // -------- double-buffered async global->LDS pipeline --------
  auto issue = [&](int buf, int k0) {
    const bf16_t* ga = A + (size_t)(blockM0 + la_row) * K + k0 + la_col;
    bf16_t* la = As[buf] + la_row * BK + la_col;
    async_b128<0>(ga, la);
    async_b128<16>(ga, la);
    const bf16_t* gb = Bt + (size_t)(blockN0 + la_row) * K + k0 + la_col;
    bf16_t* lb = Bs[buf] + la_row * BK + la_col;
    async_b128<0>(gb, lb);
    async_b128<16>(gb, lb);
  };
  issue(0, 0);
  for (int k0 = 0; k0 < K; k0 += BK) {
    int cur = (k0 >> 5) & 1;
    if (k0 + BK < K) {
      issue(cur ^ 1, k0 + BK);   // prefetch next tile into other buffer
      WAIT_ASYNC(4);             // in-order completion: current tile resident
    } else {
      WAIT_ASYNC(0);
    }
    __syncthreads();

    bf16x16 bf0 = frag_ld(Bs[cur] + (wn * 32 + 0)  * BK, BK);
    bf16x16 bf1 = frag_ld(Bs[cur] + (wn * 32 + 16) * BK, BK);
#pragma unroll
    for (int i = 0; i < 4; ++i) {
      bf16x16 af = frag_ld(As[cur] + (wm * 64 + i * 16) * BK, BK);
      acc[i][0] = wmma_bf16(af, bf0, acc[i][0]);
      acc[i][1] = wmma_bf16(af, bf1, acc[i][1]);
    }
    __syncthreads();             // protect 'cur' buffer before it is re-filled
  }
#else
  // -------- fallback: synchronous staging --------
  for (int k0 = 0; k0 < K; k0 += BK) {
    const bf16_t* ga = A + (size_t)(blockM0 + la_row) * K + k0 + la_col;
    *(bf16x8*)(As[0] + la_row * BK + la_col)     = *(const bf16x8*)(ga);
    *(bf16x8*)(As[0] + la_row * BK + la_col + 8) = *(const bf16x8*)(ga + 8);
    const bf16_t* gb = Bt + (size_t)(blockN0 + la_row) * K + k0 + la_col;
    *(bf16x8*)(Bs[0] + la_row * BK + la_col)     = *(const bf16x8*)(gb);
    *(bf16x8*)(Bs[0] + la_row * BK + la_col + 8) = *(const bf16x8*)(gb + 8);
    __syncthreads();

    bf16x16 bf0 = frag_ld(Bs[0] + (wn * 32 + 0)  * BK, BK);
    bf16x16 bf1 = frag_ld(Bs[0] + (wn * 32 + 16) * BK, BK);
#pragma unroll
    for (int i = 0; i < 4; ++i) {
      bf16x16 af = frag_ld(As[0] + (wm * 64 + i * 16) * BK, BK);
      acc[i][0] = wmma_bf16(af, bf0, acc[i][0]);
      acc[i][1] = wmma_bf16(af, bf1, acc[i][1]);
    }
    __syncthreads();
  }
#endif

  // epilogue: C layout — lane: n = lane&15, mhalf = lane>>4; VGPR v -> row v+8*mh
  int nl = lane & 15, mh = lane >> 4;
#pragma unroll
  for (int i = 0; i < 4; ++i)
#pragma unroll
    for (int j = 0; j < 2; ++j) {
      int mbase = blockM0 + wm * 64 + i * 16 + mh * 8;
      int n = blockN0 + wn * 32 + j * 16 + nl;
      float bv = bias[n];
#pragma unroll
      for (int v = 0; v < 8; ++v) {
        int m = mbase + v;
        float val = acc[i][j][v] + bv;
        size_t off = (size_t)m * N + n;
        if (mode == GM_F32) {
          outf[off] = val;
        } else if (mode == GM_RESID) {
          outf[off] = val + resid[off];
        } else if (mode == GM_BF16) {
          outb[off] = f2bf(val);
        } else if (mode == GM_GELU) {
          outb[off] = f2bf(0.5f * val * (1.0f + erff(val * 0.70710678118f)));
        } else {  // GM_QKV
          int bb = m >> 11, tt = m & (kT - 1);
          int which = n >> 10, e = n & (kE - 1);
          int hh = e >> 6, dd = e & 63;
          size_t bh = (size_t)(bb * kH + hh);
          if (which == 0)
            qo[(bh * kT + tt) * kD + dd] = f2bf(val * 0.125f);  // 1/sqrt(64)
          else if (which == 1)
            ko[(bh * kT + tt) * kD + dd] = f2bf(val);
          else
            vto[(bh * kD + dd) * kT + tt] = f2bf(val);          // V transposed
        }
      }
    }
}

// ---------- causal flash attention ----------
// q,k: [B*H, T, D] bf16 (q pre-scaled), vT: [B*H, D, T] bf16, y: [B,T,E] bf16
// block = 128 threads = 4 waves; each wave owns 16 query rows; block covers 64.

__global__ __launch_bounds__(128) void attn_kernel(const bf16_t* __restrict__ q,
                                                   const bf16_t* __restrict__ k,
                                                   const bf16_t* __restrict__ vT,
                                                   bf16_t* __restrict__ y) {
  __shared__ __align__(16) bf16_t pls[4][16 * 32];
  int wave = threadIdx.x >> 5;
  int lane = threadIdx.x & 31;
  int nl = lane & 15, mh = lane >> 4;
  int bh = blockIdx.y;                 // 0..31 (= b*16 + h)
  int qrow0 = blockIdx.x * 64 + wave * 16;

  const bf16_t* qb = q + ((size_t)bh * kT + qrow0) * kD;
  bf16x16 qf0 = frag_ld(qb + 0,  kD);    // d 0..31
  bf16x16 qf1 = frag_ld(qb + 32, kD);    // d 32..63

  v8f o0 = {}, o1 = {}, o2 = {}, o3 = {};
  float mrow[8], lrow[8];
#pragma unroll
  for (int v = 0; v < 8; ++v) { mrow[v] = -3.0e38f; lrow[v] = 0.f; }

  int kend = qrow0 + 15;   // causal bound for this wave's rows
  for (int kk = 0; kk <= kend; kk += 32) {
    const bf16_t* kptr = k + ((size_t)bh * kT + kk) * kD;
    v8f s0 = {}, s1 = {};
    {
      bf16x16 kf = frag_ld(kptr, kD);                  // keys kk..kk+15, d0..31
      s0 = wmma_bf16(qf0, kf, s0);
      kf = frag_ld(kptr + 32, kD);                     // d32..63
      s0 = wmma_bf16(qf1, kf, s0);
      kf = frag_ld(kptr + 16 * kD, kD);                // keys kk+16..kk+31
      s1 = wmma_bf16(qf0, kf, s1);
      kf = frag_ld(kptr + 16 * kD + 32, kD);
      s1 = wmma_bf16(qf1, kf, s1);
    }
    // online softmax update (rows live in VGPR index v, cols in lanes)
    bf16_t* pw = pls[wave];
#pragma unroll
    for (int v = 0; v < 8; ++v) {
      int qr = qrow0 + mh * 8 + v;
      if (kk + nl > qr)      s0[v] = -1.0e30f;
      if (kk + 16 + nl > qr) s1[v] = -1.0e30f;
      float mx = fmaxf(s0[v], s1[v]);
#pragma unroll
      for (int msk = 8; msk; msk >>= 1) mx = fmaxf(mx, __shfl_xor(mx, msk, 32));
      float nm = fmaxf(mrow[v], mx);
      float sf = __expf(mrow[v] - nm);
      mrow[v] = nm;
      float p0 = __expf(s0[v] - nm);
      float p1 = __expf(s1[v] - nm);
      float ps = p0 + p1;
#pragma unroll
      for (int msk = 8; msk; msk >>= 1) ps += __shfl_xor(ps, msk, 32);
      lrow[v] = lrow[v] * sf + ps;
      o0[v] *= sf; o1[v] *= sf; o2[v] *= sf; o3[v] *= sf;
      int r = mh * 8 + v;
      pw[r * 32 + nl]      = f2bf(p0);   // C-layout -> A-layout via LDS
      pw[r * 32 + 16 + nl] = f2bf(p1);
    }
    asm volatile("s_wait_dscnt 0" ::: "memory");
    bf16x16 pf = frag_ld(pw, 32);        // P: 16(q) x 32(keys)
    const bf16_t* vb = vT + (size_t)bh * kD * kT + kk;
    o0 = wmma_bf16(pf, frag_ld(vb + (size_t)0  * kT, kT), o0);
    o1 = wmma_bf16(pf, frag_ld(vb + (size_t)16 * kT, kT), o1);
    o2 = wmma_bf16(pf, frag_ld(vb + (size_t)32 * kT, kT), o2);
    o3 = wmma_bf16(pf, frag_ld(vb + (size_t)48 * kT, kT), o3);
  }

  int bb = bh >> 4, hh = bh & 15;
#pragma unroll
  for (int v = 0; v < 8; ++v) {
    float inv = 1.0f / lrow[v];
    int trow = qrow0 + mh * 8 + v;
    size_t base = ((size_t)(bb * kT + trow)) * kE + hh * kD;
    y[base + 0  + nl] = f2bf(o0[v] * inv);
    y[base + 16 + nl] = f2bf(o1[v] * inv);
    y[base + 32 + nl] = f2bf(o2[v] * inv);
    y[base + 48 + nl] = f2bf(o3[v] * inv);
  }
}

// ---------- launch ----------

extern "C" void kernel_launch(void* const* d_in, const int* in_sizes, int n_in,
                              void* d_out, int out_size, void* d_ws, size_t ws_size,
                              hipStream_t stream) {
  const float* x      = (const float*)d_in[0];
  const float* ln1_w  = (const float*)d_in[1];
  const float* ln1_b  = (const float*)d_in[2];
  const float* w_qkv  = (const float*)d_in[3];
  const float* b_qkv  = (const float*)d_in[4];
  const float* w_o    = (const float*)d_in[5];
  const float* b_o    = (const float*)d_in[6];
  const float* ln2_w  = (const float*)d_in[7];
  const float* ln2_b  = (const float*)d_in[8];
  const float* w_fc   = (const float*)d_in[9];
  const float* b_fc   = (const float*)d_in[10];
  const float* w_proj = (const float*)d_in[11];
  const float* b_proj = (const float*)d_in[12];
  float* outp = (float*)d_out;

  char* p = (char*)d_ws;
  auto take = [&](size_t bytes) { char* r = p; p += bytes; return r; };
  bf16_t* wqkvT = (bf16_t*)take((size_t)3072 * 1024 * 2);
  bf16_t* woT   = (bf16_t*)take((size_t)1024 * 1024 * 2);
  bf16_t* wfcT  = (bf16_t*)take((size_t)4096 * 1024 * 2);
  bf16_t* wprT  = (bf16_t*)take((size_t)1024 * 4096 * 2);
  bf16_t* h1    = (bf16_t*)take((size_t)kRows * kE * 2);
  bf16_t* qb    = (bf16_t*)take((size_t)kB * kH * kT * kD * 2);
  bf16_t* kb    = (bf16_t*)take((size_t)kB * kH * kT * kD * 2);
  bf16_t* vTb   = (bf16_t*)take((size_t)kB * kH * kT * kD * 2);
  bf16_t* yb    = (bf16_t*)take((size_t)kRows * kE * 2);
  float*  x1    = (float*)take((size_t)kRows * kE * 4);
  bf16_t* h2    = (bf16_t*)take((size_t)kRows * kE * 2);
  bf16_t* fcb   = (bf16_t*)take((size_t)kRows * 4096 * 2);

  // 1) weights -> bf16 transposed [N,K]
  convt_kernel<<<dim3(1024 / 32, 3072 / 32), 256, 0, stream>>>(w_qkv, wqkvT, 1024, 3072);
  convt_kernel<<<dim3(1024 / 32, 1024 / 32), 256, 0, stream>>>(w_o,   woT,   1024, 1024);
  convt_kernel<<<dim3(1024 / 32, 4096 / 32), 256, 0, stream>>>(w_fc,  wfcT,  1024, 4096);
  convt_kernel<<<dim3(4096 / 32, 1024 / 32), 256, 0, stream>>>(w_proj, wprT, 4096, 1024);

  // 2) LN1
  ln_bf16_kernel<<<kRows, 256, 0, stream>>>(x, ln1_w, ln1_b, h1);

  // 3) QKV GEMM (scatter epilogue, q pre-scaled by 1/sqrt(D))
  gemm_bf16_kernel<<<dim3(3072 / 128, kRows / 128), 256, 0, stream>>>(
      h1, wqkvT, b_qkv, nullptr, nullptr, nullptr, qb, kb, vTb,
      kRows, 3072, 1024, GM_QKV);

  // 4) causal flash attention
  attn_kernel<<<dim3(kT / 64, kB * kH), 128, 0, stream>>>(qb, kb, vTb, yb);

  // 5) output projection + residual -> x1 (f32)
  gemm_bf16_kernel<<<dim3(1024 / 128, kRows / 128), 256, 0, stream>>>(
      yb, woT, b_o, x, x1, nullptr, nullptr, nullptr, nullptr,
      kRows, 1024, 1024, GM_RESID);

  // 6) LN2
  ln_bf16_kernel<<<kRows, 256, 0, stream>>>(x1, ln2_w, ln2_b, h2);

  // 7) FC + exact GELU -> bf16
  gemm_bf16_kernel<<<dim3(4096 / 128, kRows / 128), 256, 0, stream>>>(
      h2, wfcT, b_fc, nullptr, nullptr, fcb, nullptr, nullptr, nullptr,
      kRows, 4096, 1024, GM_GELU);

  // 8) proj + residual(x1) -> d_out (f32)
  gemm_bf16_kernel<<<dim3(1024 / 128, kRows / 128), 256, 0, stream>>>(
      fcb, wprT, b_proj, x1, outp, nullptr, nullptr, nullptr, nullptr,
      kRows, 1024, 4096, GM_RESID);
}
